// HGTModel_22754736734950
// MI455X (gfx1250) — matrix operand order repaired and checked by reading, not compiled, not run on hardware
//
#include <hip/hip_runtime.h>
#include <hip/hip_bf16.h>
#include <math.h>

// ---------------------------------------------------------------------------
// HGT forward (2 layers, single edge type) for MI455X / gfx1250.
//   - All dense GEMMs use v_wmma_f32_16x16x32_bf16 (bf16 A/B, f32 acc).
//   - Each wave computes a 16x64 output tile (4 N-tiles): the A fragment is
//     loaded once per k-block and reused across 4 WMMAs (4x less A traffic).
//   - a_rel / m_rel per-head 16x16 transforms are folded into Wk / Wv.
//   - Edge softmax: 3-pass (atomic max, exp+atomic sum, atomic scatter) in L2.
// ---------------------------------------------------------------------------

#define N_NODES 40000
#define N_EDGES 640000
#define CDIM    128
#define HEADS   8
#define DHEAD   16
#define OUTC    64

typedef __attribute__((ext_vector_type(16))) __bf16 v16bf;
typedef __attribute__((ext_vector_type(8)))  float  v8f;

union Frag32B { uint4 u[2]; v16bf v; };

// ---------------------------------------------------------------------------
// WMMA GEMM: D[M,Nc] = A[M,K](bf16) @ Bsw(K,Nc swizzled bf16) + bias
// One 16x64 tile per wave (4 accumulators); K-loop in steps of 32.
// mode 0: store f32                (QKV projections -> Cout)
// mode 1: HGT epilogue: a=acc+bias; h=relu(beta*a+(1-beta)*h_old); also bf16
// mode 2: store f32                (final fc -> d_out)
// Requires: M % 16 == 0, K % 32 == 0, Nc % 64 == 0.
// ---------------------------------------------------------------------------
__global__ void wmma_gemm_kernel(const __hip_bfloat16* __restrict__ A,
                                 const __hip_bfloat16* __restrict__ Bsw,
                                 const float* __restrict__ bias,
                                 float* __restrict__ Cout,
                                 int M, int K, int Nc, int mode,
                                 const float* __restrict__ h_old,
                                 float* __restrict__ h_new,
                                 __hip_bfloat16* __restrict__ hbf_new,
                                 const float* __restrict__ skip_gate)
{
    const int wave = threadIdx.x >> 5;
    const int lane = threadIdx.x & 31;
    const int tile = blockIdx.x * (blockDim.x >> 5) + wave;
    const int nTilesN4 = Nc >> 6;               // groups of 4 column tiles
    const int nTiles   = (M >> 4) * nTilesN4;
    if (tile >= nTiles) return;                 // whole-wave exit: EXEC all-ones
    const int mt   = tile / nTilesN4;
    const int ct4  = tile - mt * nTilesN4;
    const int col  = lane & 15;
    const int half = lane >> 4;
    const int kbase = half << 4;                // A fragment: lanes 16-31 start at K=16
    const int KB   = K >> 5;

    const v8f vzero = {0.f, 0.f, 0.f, 0.f, 0.f, 0.f, 0.f, 0.f};
    v8f acc[4] = {vzero, vzero, vzero, vzero};

    // A: 16x32 A fragment in this code's contiguous k order (not the ISA table order): lane m holds row m;
    //    e0..15 -> K = kbase+0..15  (two 16B loads)
    const __hip_bfloat16* arow = A + (size_t)(mt * 16 + col) * K + kbase;
    // B: pre-swizzled so each lane reads its 16 elems contiguously (32B)
    const __hip_bfloat16* bbase = Bsw + ((size_t)(ct4 * 4) * KB * 32 + lane) * 16;

    for (int kb = 0; kb < KB; ++kb) {
        Frag32B af;
        af.u[0] = *reinterpret_cast<const uint4*>(arow);
        af.u[1] = *reinterpret_cast<const uint4*>(arow + 8);
#pragma unroll
        for (int n = 0; n < 4; ++n) {
            const __hip_bfloat16* bp = bbase + (size_t)(n * KB + kb) * 512;
            Frag32B bfv;
            bfv.u[0] = *reinterpret_cast<const uint4*>(bp);
            bfv.u[1] = *reinterpret_cast<const uint4*>(bp + 8);
            acc[n] = __builtin_amdgcn_wmma_f32_16x16x32_bf16(
                         false, af.v, false, bfv.v, (short)0, acc[n], false, false);
        }
        arow += 32;
    }

    // C/D layout: lane<16 -> N=lane, M=vgpr; lane>=16 -> N=lane-16, M=8+vgpr
    if (mode == 1) {
        const float beta = 1.f / (1.f + __expf(-skip_gate[0]));
        const float omb  = 1.f - beta;
#pragma unroll
        for (int n = 0; n < 4; ++n) {
            const int colg = (ct4 * 4 + n) * 16 + col;
            const float bv = bias[colg];
#pragma unroll
            for (int i = 0; i < 8; ++i) {
                const int row = mt * 16 + half * 8 + i;
                const size_t idx = (size_t)row * Nc + colg;
                float a  = acc[n][i] + bv;
                float hn = beta * a + omb * h_old[idx];
                hn = fmaxf(hn, 0.f);
                h_new[idx]   = hn;
                hbf_new[idx] = __float2bfloat16(hn);
            }
        }
    } else {
#pragma unroll
        for (int n = 0; n < 4; ++n) {
            const int colg = (ct4 * 4 + n) * 16 + col;
            const float bv = bias[colg];
#pragma unroll
            for (int i = 0; i < 8; ++i) {
                const int row = mt * 16 + half * 8 + i;
                Cout[(size_t)row * Nc + colg] = acc[n][i] + bv;
            }
        }
    }
}

// ---------------------------------------------------------------------------
// Weight prep: fuse a_rel into Wk, m_rel into Wv; emit bf16 in WMMA-B swizzled
// order. B fragment order (mirrors ISA sparse-B tables):
//   lane = col + 16*(K/16), elem e -> K = 16*(lane>>4) + e  within a k-block.
// ---------------------------------------------------------------------------
__device__ inline size_t bswz(int KB, int kRow, int j)
{
    const int ct = j >> 4, colL = j & 15;
    const int kb = kRow >> 5, kk = kRow & 31;
    const int lane = colL + ((kk >> 4) << 4);
    const int e    = kk & 15;
    return (((size_t)(ct * KB + kb)) * 32 + lane) * 16 + e;
}

__global__ void fuse_qkv_weights_kernel(const float* __restrict__ Wq,
                                        const float* __restrict__ Wk,
                                        const float* __restrict__ Wv,
                                        const float* __restrict__ a_rel,
                                        const float* __restrict__ m_rel,
                                        const float* __restrict__ bq,
                                        const float* __restrict__ bk,
                                        const float* __restrict__ bv,
                                        __hip_bfloat16* __restrict__ Bsw,
                                        float* __restrict__ biasOut)
{
    const int t = blockIdx.x * blockDim.x + threadIdx.x;
    if (t >= CDIM * 384) return;
    const int kRow = t / 384;
    const int j    = t - kRow * 384;
    float val;
    if (j < 128) {
        val = Wq[kRow * 128 + j];
    } else if (j < 256) {
        const int c = j - 128, hh = c >> 4, e = c & 15;
        float s = 0.f;
        for (int d = 0; d < 16; ++d)
            s += Wk[kRow * 128 + hh * 16 + d] * a_rel[(hh * 16 + d) * 16 + e];
        val = s;
    } else {
        const int c = j - 256, hh = c >> 4, e = c & 15;
        float s = 0.f;
        for (int d = 0; d < 16; ++d)
            s += Wv[kRow * 128 + hh * 16 + d] * m_rel[(hh * 16 + d) * 16 + e];
        val = s;
    }
    Bsw[bswz(4, kRow, j)] = __float2bfloat16(val);
    if (kRow == 0) {
        float b;
        if (j < 128) b = bq[j];
        else if (j < 256) {
            const int c = j - 128, hh = c >> 4, e = c & 15;
            float s = 0.f;
            for (int d = 0; d < 16; ++d)
                s += bk[hh * 16 + d] * a_rel[(hh * 16 + d) * 16 + e];
            b = s;
        } else {
            const int c = j - 256, hh = c >> 4, e = c & 15;
            float s = 0.f;
            for (int d = 0; d < 16; ++d)
                s += bv[hh * 16 + d] * m_rel[(hh * 16 + d) * 16 + e];
            b = s;
        }
        biasOut[j] = b;
    }
}

__global__ void swizzle_weight_kernel(const float* __restrict__ W,
                                      __hip_bfloat16* __restrict__ Bsw,
                                      int K, int Nc)
{
    const int t = blockIdx.x * blockDim.x + threadIdx.x;
    if (t >= K * Nc) return;
    const int kRow = t / Nc;
    const int j    = t - kRow * Nc;
    Bsw[bswz(K >> 5, kRow, j)] = __float2bfloat16(W[t]);
}

__global__ void f32_to_bf16_kernel(const float* __restrict__ X,
                                   __hip_bfloat16* __restrict__ Y, int n)
{
    const int t = blockIdx.x * blockDim.x + threadIdx.x;
    if (t < n) Y[t] = __float2bfloat16(X[t]);
}

__global__ void init_seg_kernel(float* __restrict__ mmax,
                                float* __restrict__ ssum,
                                float* __restrict__ agg)
{
    const int t = blockIdx.x * blockDim.x + threadIdx.x;
    if (t < N_NODES * CDIM) agg[t] = 0.f;
    if (t < N_NODES * HEADS) { mmax[t] = -INFINITY; ssum[t] = 0.f; }
}

__device__ inline void atomicMaxF(float* addr, float val)
{
    unsigned* ua = reinterpret_cast<unsigned*>(addr);
    unsigned old = *ua;
    while (__uint_as_float(old) < val) {
        const unsigned assumed = old;
        old = atomicCAS(ua, assumed, __float_as_uint(val));
        if (old == assumed) break;
    }
}

// ---- Edge pass A: logits + segment max ------------------------------------
__global__ void edge_logits_kernel(const int* __restrict__ ei,
                                   const float* __restrict__ QKV,
                                   const float* __restrict__ p_rel,
                                   float* __restrict__ logits,
                                   float* __restrict__ mmax)
{
    const int t = blockIdx.x * blockDim.x + threadIdx.x;
    if (t >= N_EDGES * HEADS) return;
    const int e = t >> 3, h = t & 7;
    const int src = ei[e];
    const int dst = ei[N_EDGES + e];
    const float4* qp = reinterpret_cast<const float4*>(QKV + (size_t)dst * 384 + h * 16);
    const float4* kp = reinterpret_cast<const float4*>(QKV + (size_t)src * 384 + 128 + h * 16);
    __builtin_prefetch(QKV + (size_t)src * 384 + 256 + h * 16, 0, 3); // V row for scatter
    float dotv = 0.f;
#pragma unroll
    for (int i = 0; i < 4; ++i) {
        const float4 q = qp[i], k = kp[i];
        dotv += q.x * k.x + q.y * k.y + q.z * k.z + q.w * k.w;
    }
    const float lg = dotv * p_rel[h] * 0.25f;   // 1/sqrt(16)
    logits[t] = lg;
    atomicMaxF(&mmax[dst * HEADS + h], lg);
}

// ---- Edge pass B: exp + segment sum ---------------------------------------
__global__ void edge_exp_kernel(const int* __restrict__ ei,
                                float* __restrict__ logits,
                                const float* __restrict__ mmax,
                                float* __restrict__ ssum)
{
    const int t = blockIdx.x * blockDim.x + threadIdx.x;
    if (t >= N_EDGES * HEADS) return;
    const int e = t >> 3, h = t & 7;
    const int dst = ei[N_EDGES + e];
    float md = mmax[dst * HEADS + h];
    if (!isfinite(md)) md = 0.f;
    const float ex = __expf(logits[t] - md);
    logits[t] = ex;
    atomicAdd(&ssum[dst * HEADS + h], ex);
}

// ---- Edge pass C: alpha * V scatter ---------------------------------------
__global__ void edge_scatter_kernel(const int* __restrict__ ei,
                                    const float* __restrict__ logits,
                                    const float* __restrict__ ssum,
                                    const float* __restrict__ QKV,
                                    float* __restrict__ agg)
{
    const int t = blockIdx.x * blockDim.x + threadIdx.x;
    if (t >= N_EDGES * HEADS) return;
    const int e = t >> 3, h = t & 7;
    const int src = ei[e];
    const int dst = ei[N_EDGES + e];
    const float alpha = logits[t] / (ssum[dst * HEADS + h] + 1e-16f);
    const float4* vp = reinterpret_cast<const float4*>(QKV + (size_t)src * 384 + 256 + h * 16);
    float* ag = agg + (size_t)dst * CDIM + h * 16;
#pragma unroll
    for (int i = 0; i < 4; ++i) {
        const float4 v = vp[i];
        atomicAdd(ag + i * 4 + 0, v.x * alpha);
        atomicAdd(ag + i * 4 + 1, v.y * alpha);
        atomicAdd(ag + i * 4 + 2, v.z * alpha);
        atomicAdd(ag + i * 4 + 3, v.w * alpha);
    }
}

// ---- gelu(agg) -> bf16 (A operand for the attention-out GEMM) -------------
__global__ void gelu_bf16_kernel(const float* __restrict__ agg,
                                 __hip_bfloat16* __restrict__ g)
{
    const int t = blockIdx.x * blockDim.x + threadIdx.x;
    if (t >= N_NODES * CDIM) return;
    const float x = agg[t];
    const float u = 0.7978845608028654f * (x + 0.044715f * x * x * x);
    g[t] = __float2bfloat16(0.5f * x * (1.f + tanhf(u)));
}

// ---------------------------------------------------------------------------
extern "C" void kernel_launch(void* const* d_in, const int* in_sizes, int n_in,
                              void* d_out, int out_size, void* d_ws, size_t ws_size,
                              hipStream_t stream)
{
    const float* x     = (const float*)d_in[0];
    const int*   ei    = (const int*)d_in[1];
    const float* Wk    = (const float*)d_in[2];
    const float* bk    = (const float*)d_in[3];
    const float* Wq    = (const float*)d_in[4];
    const float* bq    = (const float*)d_in[5];
    const float* Wv    = (const float*)d_in[6];
    const float* bv    = (const float*)d_in[7];
    const float* a_rel = (const float*)d_in[8];
    const float* m_rel = (const float*)d_in[9];
    const float* p_rel = (const float*)d_in[10];
    const float* skip  = (const float*)d_in[11];
    const float* aW    = (const float*)d_in[12];
    const float* ab    = (const float*)d_in[13];
    const float* fcW   = (const float*)d_in[14];
    const float* fcb   = (const float*)d_in[15];
    float* out = (float*)d_out;

    // ---- workspace carve (256B aligned) -----------------------------------
    char* w = (char*)d_ws;
    auto alloc = [&](size_t bytes) -> void* {
        void* p = (void*)w;
        w += (bytes + 255) & ~(size_t)255;
        return p;
    };
    __hip_bfloat16* hbf   = (__hip_bfloat16*)alloc((size_t)N_NODES * CDIM * 2);
    float*          hbuf  = (float*)alloc((size_t)N_NODES * CDIM * 4);
    float*          QKV   = (float*)alloc((size_t)N_NODES * 384 * 4);
    float*          logit = (float*)alloc((size_t)N_EDGES * HEADS * 4);
    float*          mmax  = (float*)alloc((size_t)N_NODES * HEADS * 4);
    float*          ssum  = (float*)alloc((size_t)N_NODES * HEADS * 4);
    float*          agg   = (float*)alloc((size_t)N_NODES * CDIM * 4);
    __hip_bfloat16* gbf   = (__hip_bfloat16*)alloc((size_t)N_NODES * CDIM * 2);
    __hip_bfloat16* qkvW[2]; float* qkvB[2]; __hip_bfloat16* awsw[2];
    for (int l = 0; l < 2; ++l) {
        qkvW[l] = (__hip_bfloat16*)alloc((size_t)CDIM * 384 * 2);
        qkvB[l] = (float*)alloc(384 * 4);
        awsw[l] = (__hip_bfloat16*)alloc((size_t)CDIM * CDIM * 2);
    }
    __hip_bfloat16* fcsw = (__hip_bfloat16*)alloc((size_t)CDIM * OUTC * 2);

    const int NODE_ELEMS = N_NODES * CDIM;                // 5,120,000
    const dim3 blk(256);
    const dim3 gNode((NODE_ELEMS + 255) / 256);           // 20000
    const dim3 gEdge((N_EDGES * HEADS + 255) / 256);      // 20000

    // ---- weight prep ------------------------------------------------------
    f32_to_bf16_kernel<<<gNode, blk, 0, stream>>>(x, hbf, NODE_ELEMS);
    for (int l = 0; l < 2; ++l) {
        fuse_qkv_weights_kernel<<<dim3((CDIM * 384 + 255) / 256), blk, 0, stream>>>(
            Wq + (size_t)l * CDIM * CDIM, Wk + (size_t)l * CDIM * CDIM,
            Wv + (size_t)l * CDIM * CDIM,
            a_rel + (size_t)l * HEADS * DHEAD * DHEAD,
            m_rel + (size_t)l * HEADS * DHEAD * DHEAD,
            bq + l * CDIM, bk + l * CDIM, bv + l * CDIM,
            qkvW[l], qkvB[l]);
        swizzle_weight_kernel<<<dim3((CDIM * CDIM + 255) / 256), blk, 0, stream>>>(
            aW + (size_t)l * CDIM * CDIM, awsw[l], CDIM, CDIM);
    }
    swizzle_weight_kernel<<<dim3((CDIM * OUTC + 255) / 256), blk, 0, stream>>>(
        fcW, fcsw, CDIM, OUTC);

    // ---- layers -----------------------------------------------------------
    // wave count per GEMM: (M/16) * (Nc/64); 8 waves per 256-thread block
    const int tQKV = 2500 * (384 / 64);   // 15000 waves
    const int tAW  = 2500 * (128 / 64);   //  5000 waves
    const int tFC  = 2500 * (64 / 64);    //  2500 waves

    const float* curH = x;   // layer-0 skip input (read-only)
    for (int l = 0; l < 2; ++l) {
        // QKV (with a_rel/m_rel folded): [40000,128] x [128,384]
        wmma_gemm_kernel<<<dim3((tQKV + 7) / 8), blk, 0, stream>>>(
            hbf, qkvW[l], qkvB[l], QKV, N_NODES, CDIM, 384, /*mode*/0,
            nullptr, nullptr, nullptr, nullptr);

        init_seg_kernel<<<gNode, blk, 0, stream>>>(mmax, ssum, agg);
        edge_logits_kernel<<<gEdge, blk, 0, stream>>>(ei, QKV, p_rel + l * HEADS, logit, mmax);
        edge_exp_kernel<<<gEdge, blk, 0, stream>>>(ei, logit, mmax, ssum);
        edge_scatter_kernel<<<gEdge, blk, 0, stream>>>(ei, logit, ssum, QKV, agg);
        gelu_bf16_kernel<<<gNode, blk, 0, stream>>>(agg, gbf);

        // attention-out projection + skip/relu epilogue: [40000,128]x[128,128]
        wmma_gemm_kernel<<<dim3((tAW + 7) / 8), blk, 0, stream>>>(
            gbf, awsw[l], ab + l * CDIM, nullptr, N_NODES, CDIM, CDIM, /*mode*/1,
            curH, hbuf, hbf, skip + l);
        curH = hbuf;   // layer-1 reads/writes hbuf in place (element-local)
    }

    // ---- final fc: [40000,128] x [128,64] ---------------------------------
    wmma_gemm_kernel<<<dim3((tFC + 7) / 8), blk, 0, stream>>>(
        hbf, fcsw, fcb, out, N_NODES, CDIM, OUTC, /*mode*/2,
        nullptr, nullptr, nullptr, nullptr);

    (void)in_sizes; (void)n_in; (void)out_size; (void)ws_size;
}
